// ResidualVQ_84215718740356
// MI455X (gfx1250) — matrix operand order repaired and checked by reading, not compiled, not run on hardware
//
#include <hip/hip_runtime.h>
#include <hip/hip_bf16.h>
#include <stdint.h>

typedef __attribute__((ext_vector_type(16))) __bf16 v16bf;
typedef __attribute__((ext_vector_type(8)))  float  v8f;

#define BQ 8
#define SQ 2048
#define DQ 512
#define NQ 8          // quantizer stages
#define KQ 1024       // codebook size
#define M_TOT (BQ*SQ) // 16384 rows
#define ROWS_WG 64
#define NWG (M_TOT/ROWS_WG) // 256

// ---- d_out layout (floats), tuple concatenated flat in return order
#define OUT_IDX  ((size_t)M_TOT*DQ)          // 8388608
#define OUT_EXP  (OUT_IDX + (size_t)M_TOT*NQ)// 8519680
#define OUT_LOSS (OUT_EXP + NQ)              // 8519688

// ---- ws layout: [0,16MB) bf16 hi/lo B fragments, then c2 (8192 f32)
#define WS_C2_OFF (16u*1024u*1024u)
#define FRAG_BYTES 1024u

// ---- LDS layout (dynamic)
#define LANE_STRIDE 4112u                 // 4096 used + 16 pad (bank decorrelation)
#define A_BYTES (32u*LANE_STRIDE)         // 131584
#define OFF_C2   A_BYTES                  // 4096 bytes of c2
#define OFF_KEY  (OFF_C2 + 4096u)         // 64 * u64
#define OFF_R2P  (OFF_KEY + 512u)         // 64*4 f32
#define OFF_IDX  (OFF_R2P + 1024u)        // 64 u32
#define OFF_RED  (OFF_IDX + 256u)         // 64 f32
#define SMEM_BYTES (OFF_RED + 256u)       // 137728

__device__ __forceinline__ uint32_t afrag_off(uint32_t slot, uint32_t mt, uint32_t kc, uint32_t part) {
    return slot * LANE_STRIDE + ((mt * 16u + kc) * 2u + part) * 32u;
}

struct bits32x8 { int4 a, b; };

__device__ __forceinline__ v16bf ld_frag(const uint8_t* p) {
    bits32x8 s;
    s.a = ((const int4*)p)[0];
    s.b = ((const int4*)p)[1];
    return __builtin_bit_cast(v16bf, s);
}

// split f32 -> (hi bf16 RNE, lo bf16 trunc)
__device__ __forceinline__ void split_bf16(float v, uint32_t& hb, uint32_t& lb) {
    uint32_t b = __float_as_uint(v);
    hb = (b + 0x7FFFu + ((b >> 16) & 1u)) >> 16;
    float hf = __uint_as_float(hb << 16);
    float lf = v - hf;
    lb = __float_as_uint(lf) >> 16;
}

// store 32 residual values (k-offset order kr=0..31) for row m, k-chunk kc into A-frag LDS
__device__ __forceinline__ void frag_store(uint8_t* afr, int m, int kc, const float rv[32]) {
    int mt = m >> 4, ml = m & 15;
#pragma unroll
    for (int hl = 0; hl < 2; ++hl) {
        uint32_t hw[8], lw[8];
#pragma unroll
        for (int jj = 0; jj < 8; ++jj) {
            uint32_t h2[2], l2[2];
#pragma unroll
            for (int e = 0; e < 2; ++e) {
                int j  = jj * 2 + e;
                int kr = (j < 8) ? (hl * 8 + j) : (8 + hl * 8 + j);
                split_bf16(rv[kr], h2[e], l2[e]);
            }
            hw[jj] = (h2[0] & 0xFFFFu) | (h2[1] << 16);
            lw[jj] = (l2[0] & 0xFFFFu) | (l2[1] << 16);
        }
        uint32_t slot = (uint32_t)(ml + 16 * hl);
        uint8_t* ph = afr + afrag_off(slot, mt, kc, 0);
        uint8_t* pl = afr + afrag_off(slot, mt, kc, 1);
        ((int4*)ph)[0] = make_int4((int)hw[0], (int)hw[1], (int)hw[2], (int)hw[3]);
        ((int4*)ph)[1] = make_int4((int)hw[4], (int)hw[5], (int)hw[6], (int)hw[7]);
        ((int4*)pl)[0] = make_int4((int)lw[0], (int)lw[1], (int)lw[2], (int)lw[3]);
        ((int4*)pl)[1] = make_int4((int)lw[4], (int)lw[5], (int)lw[6], (int)lw[7]);
    }
}

__device__ __forceinline__ void frag_read(const uint8_t* afr, int m, int kc, float rv[32]) {
    int mt = m >> 4, ml = m & 15;
#pragma unroll
    for (int hl = 0; hl < 2; ++hl) {
        uint32_t slot = (uint32_t)(ml + 16 * hl);
        const int4* ph = (const int4*)(afr + afrag_off(slot, mt, kc, 0));
        const int4* pl = (const int4*)(afr + afrag_off(slot, mt, kc, 1));
        int4 h0 = ph[0], h1 = ph[1], l0 = pl[0], l1 = pl[1];
        uint32_t hw[8] = {(uint32_t)h0.x,(uint32_t)h0.y,(uint32_t)h0.z,(uint32_t)h0.w,
                          (uint32_t)h1.x,(uint32_t)h1.y,(uint32_t)h1.z,(uint32_t)h1.w};
        uint32_t lw[8] = {(uint32_t)l0.x,(uint32_t)l0.y,(uint32_t)l0.z,(uint32_t)l0.w,
                          (uint32_t)l1.x,(uint32_t)l1.y,(uint32_t)l1.z,(uint32_t)l1.w};
#pragma unroll
        for (int jj = 0; jj < 8; ++jj) {
#pragma unroll
            for (int e = 0; e < 2; ++e) {
                int j  = jj * 2 + e;
                int kr = (j < 8) ? (hl * 8 + j) : (8 + hl * 8 + j);
                uint32_t hu = (hw[jj] >> (16 * e)) & 0xFFFFu;
                uint32_t lu = (lw[jj] >> (16 * e)) & 0xFFFFu;
                rv[kr] = __uint_as_float(hu << 16) + __uint_as_float(lu << 16);
            }
        }
    }
}

// --------------------------- prep kernels ---------------------------

__global__ __launch_bounds__(32) void rvq_zero_tail(float* __restrict__ out) {
    if (threadIdx.x < 16) out[OUT_EXP + threadIdx.x] = 0.0f;  // 8 expired + 8 losses
}

__global__ __launch_bounds__(256) void rvq_c2(const float* __restrict__ cb, float* __restrict__ c2) {
    int r = blockIdx.x * 256 + threadIdx.x;   // 8192 rows
    if (r >= NQ * KQ) return;
    const float4* p = (const float4*)(cb + (size_t)r * DQ);
    float s = 0.f;
    for (int i = 0; i < DQ / 4; ++i) {
        float4 v = p[i];
        s += v.x * v.x + v.y * v.y + v.z * v.z + v.w * v.w;
    }
    c2[r] = s;
}

// codebook f32 -> bf16 hi/lo B-fragments, pre-swizzled to WMMA B layout
__global__ __launch_bounds__(256) void rvq_prep_cb(const float* __restrict__ cb, uint8_t* __restrict__ ws) {
    size_t e = (size_t)blockIdx.x * 256 + threadIdx.x;   // NQ*KQ*DQ = 4194304
    if (e >= (size_t)NQ * KQ * DQ) return;
    int q = (int)(e >> 19);
    int rem = (int)(e & ((1u << 19) - 1u));
    int n = rem >> 9, k = rem & 511;
    uint32_t hb, lb;
    split_bf16(cb[e], hb, lb);
    int ntile = n >> 4, nl = n & 15, kc = k >> 5, kr = k & 31;
    int lane = nl + 16 * ((kr >> 3) & 1);   // B layout: lanes 0-15 hold k 0-7,16-23, lanes 16-31 hold k 8-15,24-31
    int j = (kr & 7) + ((kr >> 4) << 3);
    size_t base = ((((size_t)q * 64 + ntile) * 16 + kc) * 2) * FRAG_BYTES + (size_t)lane * 32 + (size_t)j * 2;
    *(uint16_t*)(ws + base)              = (uint16_t)hb;
    *(uint16_t*)(ws + base + FRAG_BYTES) = (uint16_t)lb;
}

// --------------------------- fused main kernel ---------------------------

__global__ __launch_bounds__(256) void rvq_main(const float* __restrict__ x,
                                                const float* __restrict__ cb,
                                                const uint8_t* __restrict__ ws,
                                                float* __restrict__ out) {
    extern __shared__ uint8_t smem[];
    uint8_t* afr = smem;
    float*  c2l  = (float*)(smem + OFF_C2);
    unsigned long long* keyl = (unsigned long long*)(smem + OFF_KEY);
    float*  r2p  = (float*)(smem + OFF_R2P);
    uint32_t* idxl = (uint32_t*)(smem + OFF_IDX);
    float*  red  = (float*)(smem + OFF_RED);

    const int t = threadIdx.x;
    const int wv = t >> 5, lane = t & 31;
    const int row0 = blockIdx.x * ROWS_WG;
    const int um = t >> 2, uq = t & 3;   // update/init mapping: 4 threads per row

    // ---- init: residual = x, build A frags + r2 partials
    {
        const float* xr = x + (size_t)(row0 + um) * DQ;
        float r2 = 0.f;
        for (int kc = uq * 4; kc < uq * 4 + 4; ++kc) {
            float rv[32];
            const float4* xp = (const float4*)(xr + kc * 32);
#pragma unroll
            for (int i = 0; i < 8; ++i) {
                float4 v = xp[i];
                rv[i*4+0] = v.x; rv[i*4+1] = v.y; rv[i*4+2] = v.z; rv[i*4+3] = v.w;
                r2 += v.x*v.x + v.y*v.y + v.z*v.z + v.w*v.w;
            }
            frag_store(afr, um, kc, rv);
        }
        r2p[um * 4 + uq] = r2;
    }
    __syncthreads();

    const float* c2g = (const float*)(ws + WS_C2_OFF);

    for (int q = 0; q < NQ; ++q) {
        // ---- phase A: stage c2 slice + reset argmin keys
        if (t < 64) keyl[t] = ~0ull;
        for (int i = t; i < KQ; i += 256) c2l[i] = c2g[q * KQ + i];
        __syncthreads();

        // ---- phase B: WMMA distance GEMM + per-row argmin
        {
            const int mt = wv >> 1;
            const int nbase = (wv & 1) * 32;
            const int nl = lane & 15, lh = lane >> 4;
            float best[8]; uint32_t bidx[8];
#pragma unroll
            for (int v = 0; v < 8; ++v) { best[v] = 3.4e38f; bidx[v] = 0; }

            const uint8_t* bq = ws + (size_t)q * 64 * 16 * 2 * FRAG_BYTES;
            const uint8_t* pa = afr + afrag_off((uint32_t)lane, (uint32_t)mt, 0, 0);

            for (int nt = 0; nt < 32; nt += 2) {
                int n0 = nbase + nt, n1 = n0 + 1;
                const uint8_t* pb0 = bq + (size_t)n0 * 32768 + (size_t)lane * 32;
                const uint8_t* pb1 = bq + (size_t)n1 * 32768 + (size_t)lane * 32;
                __builtin_prefetch(pb0 + 2 * 32768, 0, 3);
                __builtin_prefetch(pb1 + 2 * 32768, 0, 3);
                v8f acc0h = {}, acc0x = {}, acc1h = {}, acc1x = {};
#pragma unroll
                for (int kc = 0; kc < 16; ++kc) {
                    v16bf ah  = ld_frag(pa  + kc * 64);
                    v16bf al  = ld_frag(pa  + kc * 64 + 32);
                    v16bf b0h = ld_frag(pb0 + kc * 2048);
                    v16bf b0l = ld_frag(pb0 + kc * 2048 + 1024);
                    v16bf b1h = ld_frag(pb1 + kc * 2048);
                    v16bf b1l = ld_frag(pb1 + kc * 2048 + 1024);
                    acc0h = __builtin_amdgcn_wmma_f32_16x16x32_bf16(false, ah, false, b0h, (short)0, acc0h, false, false);
                    acc1h = __builtin_amdgcn_wmma_f32_16x16x32_bf16(false, ah, false, b1h, (short)0, acc1h, false, false);
                    acc0x = __builtin_amdgcn_wmma_f32_16x16x32_bf16(false, ah, false, b0l, (short)0, acc0x, false, false);
                    acc1x = __builtin_amdgcn_wmma_f32_16x16x32_bf16(false, ah, false, b1l, (short)0, acc1x, false, false);
                    acc0x = __builtin_amdgcn_wmma_f32_16x16x32_bf16(false, al, false, b0h, (short)0, acc0x, false, false);
                    acc1x = __builtin_amdgcn_wmma_f32_16x16x32_bf16(false, al, false, b1h, (short)0, acc1x, false, false);
                }
                float c20 = c2l[n0 * 16 + nl], c21 = c2l[n1 * 16 + nl];
                uint32_t g0 = (uint32_t)(n0 * 16 + nl), g1 = (uint32_t)(n1 * 16 + nl);
#pragma unroll
                for (int v = 0; v < 8; ++v) {
                    float d0 = fmaf(-2.f, acc0h[v] + acc0x[v], c20);
                    float d1 = fmaf(-2.f, acc1h[v] + acc1x[v], c21);
                    if (d0 < best[v]) { best[v] = d0; bidx[v] = g0; }
                    if (d1 < best[v]) { best[v] = d1; bidx[v] = g1; }
                }
            }
            // commit per-row (val,idx) via monotonic-encoded ds_min_u64
#pragma unroll
            for (int v = 0; v < 8; ++v) {
                uint32_t b = __float_as_uint(best[v]);
                uint32_t eu = (b & 0x80000000u) ? ~b : (b | 0x80000000u);
                unsigned long long key = ((unsigned long long)eu << 32) | (unsigned long long)bidx[v];
                atomicMin(&keyl[mt * 16 + lh * 8 + v], key);
            }
        }
        __syncthreads();

        // ---- phase C: indices + loss terms
        if (t < 64) {
            unsigned long long key = keyl[t];
            uint32_t idx = (uint32_t)key;
            uint32_t eu  = (uint32_t)(key >> 32);
            uint32_t fb  = (eu & 0x80000000u) ? (eu ^ 0x80000000u) : ~eu;
            float minv = __uint_as_float(fb);
            float r2 = r2p[t*4] + r2p[t*4+1] + r2p[t*4+2] + r2p[t*4+3];
            red[t] = r2 + minv;                       // = min ||r - c||^2 for this row
            idxl[t] = idx;
            out[OUT_IDX + (size_t)(row0 + t) * NQ + q] = (float)idx;
        }
        __syncthreads();
        if (t == 0) {
            float s = 0.f;
            for (int i = 0; i < 64; ++i) s += red[i];
            atomicAdd(out + OUT_LOSS + q, s * (1.0f / (float)((size_t)M_TOT * DQ)));
        }

        // ---- phase D: residual -= codebook[idx]; refresh frags + r2 partials
        {
            uint32_t idx = idxl[um];
            const float* crow = cb + ((size_t)q * KQ + idx) * DQ;
            float r2 = 0.f;
            for (int kc = uq * 4; kc < uq * 4 + 4; ++kc) {
                float rv[32];
                frag_read(afr, um, kc, rv);
                const float4* cp = (const float4*)(crow + kc * 32);
#pragma unroll
                for (int i = 0; i < 8; ++i) {
                    float4 c4 = cp[i];
                    rv[i*4+0] -= c4.x; rv[i*4+1] -= c4.y; rv[i*4+2] -= c4.z; rv[i*4+3] -= c4.w;
                    r2 += rv[i*4+0]*rv[i*4+0] + rv[i*4+1]*rv[i*4+1]
                        + rv[i*4+2]*rv[i*4+2] + rv[i*4+3]*rv[i*4+3];
                }
                frag_store(afr, um, kc, rv);
            }
            r2p[um * 4 + uq] = r2;
        }
        __syncthreads();
    }

    // ---- output: quantized_out = x - residual_final
    {
        const float* xr = x + (size_t)(row0 + um) * DQ;
        float* orow = out + (size_t)(row0 + um) * DQ;
        for (int kc = uq * 4; kc < uq * 4 + 4; ++kc) {
            float rv[32];
            frag_read(afr, um, kc, rv);
            const float4* xp = (const float4*)(xr + kc * 32);
            float4* op = (float4*)(orow + kc * 32);
#pragma unroll
            for (int i = 0; i < 8; ++i) {
                float4 v = xp[i];
                float4 o;
                o.x = v.x - rv[i*4+0]; o.y = v.y - rv[i*4+1];
                o.z = v.z - rv[i*4+2]; o.w = v.w - rv[i*4+3];
                op[i] = o;
            }
        }
    }
}

// --------------------------- launch ---------------------------

extern "C" void kernel_launch(void* const* d_in, const int* in_sizes, int n_in,
                              void* d_out, int out_size, void* d_ws, size_t ws_size,
                              hipStream_t stream) {
    const float* x  = (const float*)d_in[0];
    const float* cb = (const float*)d_in[1];
    float* out = (float*)d_out;
    uint8_t* ws = (uint8_t*)d_ws;

    (void)in_sizes; (void)n_in; (void)out_size; (void)ws_size;

    hipFuncSetAttribute((const void*)rvq_main,
                        hipFuncAttributeMaxDynamicSharedMemorySize, (int)SMEM_BYTES);

    rvq_zero_tail<<<1, 32, 0, stream>>>(out);
    rvq_c2<<<(NQ * KQ + 255) / 256, 256, 0, stream>>>(cb, (float*)(ws + WS_C2_OFF));
    rvq_prep_cb<<<((size_t)NQ * KQ * DQ) / 256, 256, 0, stream>>>(cb, ws);
    rvq_main<<<NWG, 256, SMEM_BYTES, stream>>>(x, cb, ws, out);
}
